// FFHeavyPath_40931038331142
// MI455X (gfx1250) — matrix operand order repaired and checked by reading, not compiled, not run on hardware
//
#include <hip/hip_runtime.h>

typedef __attribute__((ext_vector_type(16))) _Float16 v16h;
typedef __attribute__((ext_vector_type(8)))  float    v8f;
typedef __attribute__((ext_vector_type(4)))  unsigned int u32x4;
typedef __attribute__((ext_vector_type(8)))  int          i32x8;
typedef __attribute__((ext_vector_type(4)))  int          i32x4;

#define CC 256
#define TT 16
#define KK 128
#define HH 256
#define WWI 256
#define XPITCH 264   // padded LDS row pitch (halves): 528B -> bank stride 4

#if defined(__has_builtin)
#  if __has_builtin(__builtin_amdgcn_tensor_load_to_lds)
#    define HAVE_TDM 1
#  endif
#endif

__device__ __forceinline__ float sigmoidf_(float x) { return 1.f / (1.f + __expf(-x)); }
__device__ __forceinline__ float softplusf_(float x) {
  return (x > 0.f) ? (x + log1pf(__expf(-x))) : log1pf(__expf(x));
}
__device__ __forceinline__ float gelu_tanh_(float x) {
  float u = 0.7978845608028654f * (x + 0.044715f * x * x * x);
  return 0.5f * x * (1.f + tanhf(u));
}

// ---------------- inverse tile map: inv[b][tilePos] = slot or -1 ----------------
__global__ __launch_bounds__(1024) void invmap_kernel(const int* __restrict__ tidx,
                                                      int* __restrict__ inv) {
  int t = threadIdx.x;               // 0..1023 = B*256
  inv[t] = -1;
  __syncthreads();
  if (t < 4 * KK) {
    int b = t >> 7;
    inv[b * 256 + tidx[t]] = t & 127;
  }
}

// ---------------- prepack conv weights: [co][ci][3][3] f32 -> [tap][ci][co] f16 ----
__global__ __launch_bounds__(256) void wpack_kernel(const float* __restrict__ w1,
                                                    const float* __restrict__ w2,
                                                    _Float16* __restrict__ p1,
                                                    _Float16* __restrict__ p2) {
  int tap = blockIdx.x >> 8;
  int ci  = blockIdx.x & 255;
  int co  = threadIdx.x;
  int dst = (tap * 256 + ci) * 256 + co;
  int src = (co * 256 + ci) * 9 + tap;
  p1[dst] = (_Float16)w1[src];
  p2[dst] = (_Float16)w2[src];
}

// ---------------- tokens: per-(b,slot,c) mean over 16x16 tile ----------------
__global__ __launch_bounds__(256) void tokens_kernel(const float* __restrict__ dense,
                                                     const int* __restrict__ tidx,
                                                     float* __restrict__ tokens) {
  int bk = blockIdx.x;                     // b*128+slot
  int b = bk >> 7;
  int tile = tidx[bk];
  int gy0 = (tile >> 4) * TT, gx0 = (tile & 15) * TT;
  int c = threadIdx.x;
  const float* dch = dense + ((size_t)(b * CC + c) * HH + gy0) * WWI + gx0;
  float s = 0.f;
  for (int r = 0; r < TT; ++r) {
    const float4* row = (const float4*)(dch + r * WWI);
    for (int q = 0; q < 4; ++q) { float4 v = row[q]; s += v.x + v.y + v.z + v.w; }
  }
  tokens[bk * CC + c] = s * (1.f / 256.f);
}

// ---------------- bidirectional diagonal SSM over slots ----------------
__global__ __launch_bounds__(256) void ssm_kernel(
    const float* __restrict__ tokens,
    const float* __restrict__ decay_f, const float* __restrict__ igain_f,
    const float* __restrict__ ogain_f, const float* __restrict__ sbias_f,
    const float* __restrict__ decay_b, const float* __restrict__ igain_b,
    const float* __restrict__ ogain_b, const float* __restrict__ sbias_b,
    const float* __restrict__ merge_gate,
    float* __restrict__ mixed, float* __restrict__ state) {
  int b = blockIdx.x, c = threadIdx.x;
  float dF = sigmoidf_(decay_f[c]), iF = softplusf_(igain_f[c]);
  float oF = softplusf_(ogain_f[c]), sF = sbias_f[c];
  float dB = sigmoidf_(decay_b[c]), iB = softplusf_(igain_b[c]);
  float oB = softplusf_(ogain_b[c]), sB = sbias_b[c];
  float g = sigmoidf_(merge_gate[c]);
  float s = 0.f;
  for (int k = 0; k < KK; ++k) {
    float x = tokens[(b * KK + k) * CC + c];
    s = dF * s + iF * x + sF;
    mixed[(b * KK + k) * CC + c] = g * (oF * s);
  }
  state[(b * 2 + 0) * CC + c] = s;
  s = 0.f;
  for (int k = KK - 1; k >= 0; --k) {
    float x = tokens[(b * KK + k) * CC + c];
    s = dB * s + iB * x + sB;
    mixed[(b * KK + k) * CC + c] += (1.f - g) * (oB * s);
  }
  state[(b * 2 + 1) * CC + c] = s;
}

// ---------------- f32 -> f16 conversions for the WMMA GEMMs ----------------
__global__ __launch_bounds__(256) void cvt_kernel(
    const float* __restrict__ mixed, const float* __restrict__ Wg,
    const float* __restrict__ Wb, _Float16* __restrict__ mixed16,
    _Float16* __restrict__ Wg16, _Float16* __restrict__ Wb16) {
  int i = blockIdx.x * 256 + threadIdx.x;
  if (i < 512 * CC) mixed16[i] = (_Float16)mixed[i];
  if (i < CC * CC) { Wg16[i] = (_Float16)Wg[i]; Wb16[i] = (_Float16)Wb[i]; }
}

// ---------------- FiLM GEMMs (512x256 @ 256x256, x2) with WMMA ----------------
__global__ __launch_bounds__(128) void film_kernel(
    const _Float16* __restrict__ mixed16, const _Float16* __restrict__ Wg16,
    const _Float16* __restrict__ Wb16, const float* __restrict__ bg,
    const float* __restrict__ bb, float* __restrict__ gammaO,
    float* __restrict__ betaO) {
  __shared__ _Float16 arows[16 * 256];     // 8 KB A-panel
  const int tid = threadIdx.x, lane = tid & 31, w = tid >> 5;
  const int mt = blockIdx.x;                  // 0..31
  const int ml = lane & 15, kh = lane >> 4;

#if defined(HAVE_TDM)
  // Stage the contiguous 8KB A-panel with the Tensor Data Mover.
  if (w == 0) {
    unsigned long long ga =
        (unsigned long long)(uintptr_t)(mixed16 + (size_t)mt * 16 * 256);
    unsigned int lds = (unsigned int)(uintptr_t)&arows[0];
    u32x4 g0 = { 1u,                                  // count=1, user descriptor
                 lds,                                 // lds_addr (bytes)
                 (unsigned int)ga,                    // global_addr[31:0]
                 (unsigned int)(ga >> 32) | 0x80000000u };  // addr[56:32] | type=2
    i32x8 g1 = { 0x00010000,                          // data_size=2B, wg_mask=0
                 (int)(4096u << 16),                  // tensor_dim0[15:0]=4096
                 0x00010000,                          // tensor_dim1=1
                 (int)(4096u << 16),                  // tile_dim0=4096
                 1,                                   // tile_dim1=1
                 4096, 0, 0 };                        // tensor_dim0_stride=4096
    i32x4 z4 = { 0, 0, 0, 0 };
#if __clang_major__ >= 23
    i32x8 z8 = { 0, 0, 0, 0, 0, 0, 0, 0 };
    __builtin_amdgcn_tensor_load_to_lds(g0, g1, z4, z4, z8, 0);
#else
    __builtin_amdgcn_tensor_load_to_lds(g0, g1, z4, z4, 0);
#endif
    __builtin_amdgcn_s_wait_tensorcnt(0);
  }
#else
  for (int i = tid; i < 512; i += 128)
    ((uint4*)arows)[i] = ((const uint4*)(mixed16 + (size_t)mt * 16 * 256))[i];
#endif
  __syncthreads();

  v8f ag[4], ab[4];
  #pragma unroll
  for (int t = 0; t < 4; ++t)
    #pragma unroll
    for (int v = 0; v < 8; ++v) { ag[t][v] = 0.f; ab[t][v] = 0.f; }

  #pragma unroll 1
  for (int cb = 0; cb < 8; ++cb) {
    const int cbase = cb * 32;
    v16h A;
    const _Float16* ap = &arows[ml * 256 + cbase + kh * 8];
    #pragma unroll
    for (int j = 0; j < 8; ++j) { A[j] = ap[j]; A[j + 8] = ap[16 + j]; }
    #pragma unroll
    for (int t = 0; t < 4; ++t) {
      const int nb = (w * 4 + t) * 16;
      v16h Bg, Bb;
      #pragma unroll
      for (int j = 0; j < 8; ++j) {
        Bg[j]     = Wg16[(size_t)(cbase + kh * 8 + j) * 256 + nb + ml];
        Bg[j + 8] = Wg16[(size_t)(cbase + 16 + kh * 8 + j) * 256 + nb + ml];
        Bb[j]     = Wb16[(size_t)(cbase + kh * 8 + j) * 256 + nb + ml];
        Bb[j + 8] = Wb16[(size_t)(cbase + 16 + kh * 8 + j) * 256 + nb + ml];
      }
      ag[t] = __builtin_amdgcn_wmma_f32_16x16x32_f16(false, A, false, Bg, (short)0,
                                                     ag[t], false, false);
      ab[t] = __builtin_amdgcn_wmma_f32_16x16x32_f16(false, A, false, Bb, (short)0,
                                                     ab[t], false, false);
    }
  }
  #pragma unroll
  for (int t = 0; t < 4; ++t) {
    const int n = (w * 4 + t) * 16 + ml;
    const float bgn = bg[n], bbn = bb[n];
    #pragma unroll
    for (int v = 0; v < 8; ++v) {
      const int r = mt * 16 + v + 8 * kh;
      gammaO[r * CC + n] = tanhf(ag[t][v] + bgn);   // GAMMA_LIMIT = 1
      betaO[r * CC + n]  = ab[t][v] + bbn;
    }
  }
}

// ---------------- per-tile: modulate -> conv1 -> GN -> GELU -> conv2 -> residual+fuse
__global__ __launch_bounds__(512) void conv_refine_kernel(
    const float* __restrict__ dense, const int* __restrict__ tidx,
    const float* __restrict__ bprox, const float* __restrict__ uprox,
    const float* __restrict__ gammaO, const float* __restrict__ betaO,
    const _Float16* __restrict__ wpk1, const _Float16* __restrict__ wpk2,
    const float* __restrict__ c1b, const float* __restrict__ c2b,
    const float* __restrict__ gnsc, const float* __restrict__ gnbi,
    const float* __restrict__ fusew, const float* __restrict__ fuseb,
    float* __restrict__ heavy, float* __restrict__ detail) {
  __shared__ _Float16 xbuf[256 * XPITCH];  // [pixel][ci], k-major, padded: ~132 KB
  __shared__ float alph[256];
  __shared__ float red[2];

  const int tid = threadIdx.x;
  const int lane = tid & 31;
  const int wave = tid >> 5;               // 16 waves, 4x4 grid of 64x64 blocks
  const int wm = wave >> 2, wn = wave & 3;
  const int ml = lane & 15, kh = lane >> 4;
  const int bk = blockIdx.x;               // b*128+slot
  const int b = bk >> 7;
  const int tile = tidx[bk];
  const int gy0 = (tile >> 4) * TT, gx0 = (tile & 15) * TT;

  if (tid < 2) red[tid] = 0.f;
  if (tid < 256) {
    int gidx = (b * HH + gy0 + (tid >> 4)) * WWI + gx0 + (tid & 15);
    alph[tid] = sigmoidf_(fusew[0] * bprox[gidx] + fusew[1] * uprox[gidx] + fuseb[0]);
  }
  { // gather + FiLM modulate -> xbuf[pixel][ci] (f16)
    const int ci = tid >> 1;
    const int p0 = (tid & 1) * 128;
    const float sc = 1.f + gammaO[bk * CC + ci];
    const float bt = betaO[bk * CC + ci];
    const float* dch = dense + ((size_t)(b * CC + ci) * HH + gy0) * WWI + gx0;
    for (int r = 0; r < 8; ++r) {
      const int py = (p0 >> 4) + r;
      const float4* row = (const float4*)(dch + py * WWI);
      for (int q = 0; q < 4; ++q) {
        float4 v = row[q];
        int p = py * 16 + q * 4;
        xbuf[(p + 0) * XPITCH + ci] = (_Float16)(v.x * sc + bt);
        xbuf[(p + 1) * XPITCH + ci] = (_Float16)(v.y * sc + bt);
        xbuf[(p + 2) * XPITCH + ci] = (_Float16)(v.z * sc + bt);
        xbuf[(p + 3) * XPITCH + ci] = (_Float16)(v.w * sc + bt);
      }
    }
  }
  __syncthreads();

  v8f acc[4][4];

  auto conv_pass = [&](const _Float16* __restrict__ wglob, const float* __restrict__ cbias) {
    #pragma unroll
    for (int i = 0; i < 4; ++i)
      #pragma unroll
      for (int t = 0; t < 4; ++t)
        #pragma unroll
        for (int v = 0; v < 8; ++v) acc[i][t][v] = 0.f;
    #pragma unroll 1
    for (int tap = 0; tap < 9; ++tap) {
      const int dy = tap / 3 - 1, dx = tap % 3 - 1;
      if (tap + 1 < 9)  // prefetch next tap's weight panel (global_prefetch_b8)
        __builtin_prefetch(wglob + (size_t)(tap + 1) * 65536 + (size_t)tid * 128, 0, 1);
      int sp[4]; bool val[4];
      #pragma unroll
      for (int i = 0; i < 4; ++i) {
        int p = (wm * 4 + i) * 16 + ml;          // output pixel (A row)
        int py = (p >> 4) + dy, px = (p & 15) + dx;
        val[i] = (py >= 0) && (py < 16) && (px >= 0) && (px < 16);
        sp[i] = py * 16 + px;
      }
      const _Float16* wtap = wglob + (size_t)tap * 65536;
      #pragma unroll 1
      for (int cb = 0; cb < 8; ++cb) {
        const int cbase = cb * 32;
        v16h A[4], Bv[4];
        #pragma unroll
        for (int i = 0; i < 4; ++i) {
          if (val[i]) {
            const _Float16* ap = &xbuf[sp[i] * XPITCH + cbase + kh * 8];
            #pragma unroll
            for (int j = 0; j < 8; ++j) { A[i][j] = ap[j]; A[i][j + 8] = ap[16 + j]; }
          } else {
            #pragma unroll
            for (int j = 0; j < 16; ++j) A[i][j] = (_Float16)0.f;
          }
        }
        #pragma unroll
        for (int t = 0; t < 4; ++t)
          #pragma unroll
          for (int j = 0; j < 8; ++j) {
            Bv[t][j]     = wtap[(size_t)(cbase + kh * 8 + j) * 256 + (wn * 4 + t) * 16 + ml];
            Bv[t][j + 8] = wtap[(size_t)(cbase + 16 + kh * 8 + j) * 256 + (wn * 4 + t) * 16 + ml];
          }
        #pragma unroll
        for (int t = 0; t < 4; ++t)
          #pragma unroll
          for (int i = 0; i < 4; ++i)
            acc[i][t] = __builtin_amdgcn_wmma_f32_16x16x32_f16(
                false, A[i], false, Bv[t], (short)0, acc[i][t], false, false);
      }
    }
    #pragma unroll
    for (int t = 0; t < 4; ++t) {
      const float cbv = cbias[(wn * 4 + t) * 16 + ml];
      #pragma unroll
      for (int i = 0; i < 4; ++i)
        #pragma unroll
        for (int v = 0; v < 8; ++v) acc[i][t][v] += cbv;
    }
  };

  // ---- conv1 ----
  conv_pass(wpk1, c1b);
  __syncthreads();                          // all waves done reading xbuf

  // GN stats from f32 accumulators; write h (f16) into xbuf[pixel][co]
  float s1 = 0.f, s2 = 0.f;
  #pragma unroll
  for (int i = 0; i < 4; ++i)
    #pragma unroll
    for (int t = 0; t < 4; ++t) {
      const int co = (wn * 4 + t) * 16 + ml;
      #pragma unroll
      for (int v = 0; v < 8; ++v) {
        float h = acc[i][t][v];
        s1 += h; s2 += h * h;
        int m = (wm * 4 + i) * 16 + v + 8 * kh;
        xbuf[m * XPITCH + co] = (_Float16)h;
      }
    }
  for (int off = 16; off > 0; off >>= 1) {
    s1 += __shfl_down(s1, off, 32);
    s2 += __shfl_down(s2, off, 32);
  }
  if (lane == 0) { atomicAdd(&red[0], s1); atomicAdd(&red[1], s2); }
  __syncthreads();
  const float mu = red[0] * (1.f / 65536.f);
  const float var = red[1] * (1.f / 65536.f) - mu * mu;
  const float rs = rsqrtf(var + 1e-5f);
  { // normalize + GELU in place
    const int ci = tid >> 1;
    const int p0 = (tid & 1) * 128;
    const float sc = gnsc[ci] * rs;
    const float bi = gnbi[ci] - sc * mu;
    for (int pp = 0; pp < 128; ++pp) {
      float h = (float)xbuf[(p0 + pp) * XPITCH + ci];
      xbuf[(p0 + pp) * XPITCH + ci] = (_Float16)gelu_tanh_(sc * h + bi);
    }
  }
  __syncthreads();

  // ---- conv2 ----
  conv_pass(wpk2, c2b);

  // ---- residual (recompute modulated x in f32) + fusion epilogue ----
  #pragma unroll
  for (int t = 0; t < 4; ++t) {
    const int co = (wn * 4 + t) * 16 + ml;
    const float sc = 1.f + gammaO[bk * CC + co];
    const float bt = betaO[bk * CC + co];
    #pragma unroll
    for (int i = 0; i < 4; ++i) {
      #pragma unroll
      for (int v = 0; v < 8; ++v) {
        const int m = (wm * 4 + i) * 16 + v + 8 * kh;
        const size_t gidx =
            ((size_t)(b * CC + co) * HH + gy0 + (m >> 4)) * WWI + gx0 + (m & 15);
        const float d = dense[gidx];
        const float refined = (d * sc + bt) + acc[i][t][v];
        const float a = alph[m];
        heavy[gidx]  = a * refined + (1.f - a) * d;
        detail[gidx] = a * (refined - d);
      }
    }
  }
}

// ---------------- full image: alpha output + passthrough for unselected tiles ----
__global__ __launch_bounds__(256) void fuse_kernel(
    const float* __restrict__ dense, const float* __restrict__ bprox,
    const float* __restrict__ uprox, const int* __restrict__ inv,
    const float* __restrict__ fusew, const float* __restrict__ fuseb,
    float* __restrict__ heavy, float* __restrict__ detail,
    float* __restrict__ alpha) {
  int i = blockIdx.x * 256 + threadIdx.x;        // 0 .. 4*65536-1
  int b = i >> 16, p = i & 65535;
  int gy = p >> 8, gx = p & 255;
  float a = sigmoidf_(fusew[0] * bprox[i] + fusew[1] * uprox[i] + fuseb[0]);
  alpha[i] = a;
  int tile = (gy >> 4) * 16 + (gx >> 4);
  if (inv[b * 256 + tile] < 0) {
    for (int c = 0; c < CC; ++c) {
      size_t idx = ((size_t)(b * CC + c) << 16) + p;
      heavy[idx] = dense[idx];
      detail[idx] = 0.f;
    }
  }
}

extern "C" void kernel_launch(void* const* d_in, const int* in_sizes, int n_in,
                              void* d_out, int out_size, void* d_ws, size_t ws_size,
                              hipStream_t stream) {
  (void)in_sizes; (void)n_in; (void)out_size; (void)ws_size;
  const float* dense   = (const float*)d_in[0];
  const int*   tidx    = (const int*)d_in[1];
  const float* bprox   = (const float*)d_in[2];
  const float* uprox   = (const float*)d_in[3];
  const float* decay_f = (const float*)d_in[4];
  const float* igain_f = (const float*)d_in[5];
  const float* ogain_f = (const float*)d_in[6];
  const float* sbias_f = (const float*)d_in[7];
  const float* decay_b = (const float*)d_in[8];
  const float* igain_b = (const float*)d_in[9];
  const float* ogain_b = (const float*)d_in[10];
  const float* sbias_b = (const float*)d_in[11];
  const float* mgate   = (const float*)d_in[12];
  const float* Wg      = (const float*)d_in[13];
  const float* bg      = (const float*)d_in[14];
  const float* Wb      = (const float*)d_in[15];
  const float* bb      = (const float*)d_in[16];
  const float* conv1_w = (const float*)d_in[17];
  const float* conv1_b = (const float*)d_in[18];
  const float* gn_sc   = (const float*)d_in[19];
  const float* gn_bi   = (const float*)d_in[20];
  const float* conv2_w = (const float*)d_in[21];
  const float* conv2_b = (const float*)d_in[22];
  const float* fuse_w  = (const float*)d_in[23];
  const float* fuse_b  = (const float*)d_in[24];

  float* out = (float*)d_out;
  float* heavyO  = out;
  float* detailO = out + 67108864ll;
  float* alphaO  = out + 134217728ll;
  float* tokensO = out + 134479872ll;
  float* mixedO  = out + 134610944ll;
  float* gammaO  = out + 134742016ll;
  float* betaO   = out + 134873088ll;
  float* stateO  = out + 135004160ll;

  char* ws = (char*)d_ws;
  int*       inv     = (int*)ws;                     //   4 KB
  _Float16*  mixed16 = (_Float16*)(ws + 4096);       // 256 KB
  _Float16*  Wg16    = (_Float16*)(ws + 266240);     // 128 KB
  _Float16*  Wb16    = (_Float16*)(ws + 397312);     // 128 KB
  _Float16*  wpk1    = (_Float16*)(ws + 528384);     // 1.125 MB
  _Float16*  wpk2    = (_Float16*)(ws + 1708032);    // 1.125 MB

  invmap_kernel<<<1, 1024, 0, stream>>>(tidx, inv);
  wpack_kernel<<<2304, 256, 0, stream>>>(conv1_w, conv2_w, wpk1, wpk2);
  tokens_kernel<<<512, 256, 0, stream>>>(dense, tidx, tokensO);
  ssm_kernel<<<4, 256, 0, stream>>>(tokensO, decay_f, igain_f, ogain_f, sbias_f,
                                    decay_b, igain_b, ogain_b, sbias_b, mgate,
                                    mixedO, stateO);
  cvt_kernel<<<512, 256, 0, stream>>>(mixedO, Wg, Wb, mixed16, Wg16, Wb16);
  film_kernel<<<32, 128, 0, stream>>>(mixed16, Wg16, Wb16, bg, bb, gammaO, betaO);
  conv_refine_kernel<<<512, 512, 0, stream>>>(dense, tidx, bprox, uprox, gammaO, betaO,
                                              wpk1, wpk2, conv1_b, conv2_b, gn_sc, gn_bi,
                                              fuse_w, fuse_b, heavyO, detailO);
  fuse_kernel<<<1024, 256, 0, stream>>>(dense, bprox, uprox, inv, fuse_w, fuse_b,
                                        heavyO, detailO, alphaO);
}
